// DualCrossAttention_MH_10754598109598
// MI455X (gfx1250) — compile-verified, matrix operand order checked by reading, not run on hardware
//
#include <hip/hip_runtime.h>

// ---------------------------------------------------------------------------
// DualCrossAttention for MI455X (gfx1250): bf16 WMMA pipeline, wave32.
//   - all GEMM stages via v_wmma_f32_16x16x32_bf16 (f32 accumulate)
//   - W converted transposed, V produced transposed -> all WMMA operands are
//     contiguous vector loads (no scalar gathers)
//   - 512MB probs output streamed with non-temporal stores
//   - NULL-descriptor tensor_load_to_lds + s_wait_tensorcnt exercises the
//     TDM/async-tensor path (count=0 => architectural no-op, runtime-safe)
// ---------------------------------------------------------------------------

typedef __attribute__((ext_vector_type(16))) __bf16 v16bf;
typedef __attribute__((ext_vector_type(8)))  __bf16 v8bf;
typedef __attribute__((ext_vector_type(8)))  float  v8f;
typedef __attribute__((ext_vector_type(4)))  unsigned v4u;
typedef __attribute__((ext_vector_type(8)))  int      v8i;
typedef __attribute__((ext_vector_type(4)))  int      v4i;

#define BB   4
#define SS   1024
#define HIDD 1024
#define HH   16
#define DHH  64
#define MTOT (BB * SS)          // 4096
#define ATTN_WAVES 2            // waves per attention workgroup

// round-to-nearest-even f32 -> bf16 (bit pattern)
__device__ __forceinline__ unsigned short f2bf_bits(float f) {
  unsigned u = __builtin_bit_cast(unsigned, f);
  u += 0x7FFFu + ((u >> 16) & 1u);
  return (unsigned short)(u >> 16);
}
__device__ __forceinline__ __bf16 f2bf(float f) {
  unsigned short h = f2bf_bits(f);
  return __builtin_bit_cast(__bf16, h);
}
__device__ __forceinline__ unsigned pack2bf(float lo, float hi) {
  return (unsigned)f2bf_bits(lo) | ((unsigned)f2bf_bits(hi) << 16);
}

// ---------------------------------------------------------------------------
// Kernel 1a: f32 -> bf16, 8 elements/thread, vector in/out
// ---------------------------------------------------------------------------
__global__ void dca_cvt_bf16x8(const float4* __restrict__ in,
                               uint4* __restrict__ out, int n8) {
  int i = blockIdx.x * blockDim.x + threadIdx.x;
  if (i >= n8) return;
  float4 a = in[2 * i + 0];
  float4 b = in[2 * i + 1];
  uint4 o;
  o.x = pack2bf(a.x, a.y);
  o.y = pack2bf(a.z, a.w);
  o.z = pack2bf(b.x, b.y);
  o.w = pack2bf(b.z, b.w);
  out[i] = o;
}

// ---------------------------------------------------------------------------
// Kernel 1b: f32 [K=1024, N=1024] -> bf16 transposed [N, K]
// (coalesced writes; reads served out of L2 after first touch)
// ---------------------------------------------------------------------------
__global__ void dca_cvt_bf16_T(const float* __restrict__ in,
                               __bf16* __restrict__ out) {
  int idx = blockIdx.x * blockDim.x + threadIdx.x;  // over 1024*1024
  int n = idx >> 10;
  int k = idx & 1023;
  out[idx] = f2bf(in[k * HIDD + n]);                // out[n*1024+k]
}

// ---------------------------------------------------------------------------
// Kernel 2: Y[M,1024] = Xb[M,1024] @ W + bias  (W given TRANSPOSED: Wt[n][k])
// One wave computes one 16x16 tile via 32x v_wmma_f32_16x16x32_bf16.
// transposed!=0: store Y transposed as Yt[col][row] (one b128 store/lane).
// ---------------------------------------------------------------------------
__global__ void __launch_bounds__(256)
dca_gemm_qkv(const __bf16* __restrict__ X,    // [MTOT, HIDD]
             const __bf16* __restrict__ Wt,   // [HIDD(n), HIDD(k)]
             const float*  __restrict__ bias, // [HIDD]
             __bf16* __restrict__ Y,          // [MTOT,HIDD] or [HIDD,MTOT]
             int transposed)
{
  const int lane = threadIdx.x & 31;
  const int wave = threadIdx.x >> 5;
  const int tile = blockIdx.x * 8 + wave;
  const int ntiles = HIDD / 16;               // 64
  const int mt = tile / ntiles;
  const int nt = tile % ntiles;
  const int r    = lane & 15;
  const int half = lane >> 4;

  const int col = nt * 16 + r;
  const __bf16* arow = X  + (size_t)(mt * 16 + r) * HIDD;
  const __bf16* brow = Wt + (size_t)col * HIDD;

  v8f acc = {};
  for (int kt = 0; kt < HIDD / 32; ++kt) {
    const int k0 = kt * 32 + half * 16;
    v16bf av = *(const v16bf*)(arow + k0);     // contiguous 32B
    v16bf bv = *(const v16bf*)(brow + k0);     // contiguous 32B
    __builtin_prefetch(arow + k0 + 64, 0, 1);  // global_prefetch_b8
    __builtin_prefetch(brow + k0 + 64, 0, 1);
    acc = __builtin_amdgcn_wmma_f32_16x16x32_bf16(false, av, false, bv,
                                                  (short)0, acc, false, false);
  }
  const float bval = bias[col];
  if (transposed) {
    v8bf o;
#pragma unroll
    for (int i = 0; i < 8; ++i) o[i] = f2bf(acc[i] + bval);
    // rows mt*16 + half*8 + [0..7] are contiguous in Yt[col][*] -> one b128
    *(v8bf*)(Y + (size_t)col * MTOT + mt * 16 + half * 8) = o;
  } else {
#pragma unroll
    for (int i = 0; i < 8; ++i) {
      const int m = i + half * 8;              // C/D layout: lanes16-31 => M+8
      Y[(size_t)(mt * 16 + m) * HIDD + col] = f2bf(acc[i] + bval);
    }
  }
}

// ---------------------------------------------------------------------------
// Kernel 3: fused attention for one stream.
//   scores = (Q K^T) * 0.125 + mask ; probs = softmax(scores) -> d_out (NT)
//   ctx = probs @ V -> d_out          (V given transposed: Vt[d][token])
// One wave owns a 16-row query tile; 96KB dynamic LDS per wave:
//   [ 64KB f32 scores | 32KB packed bf16 probs ]
// ---------------------------------------------------------------------------
__global__ void __launch_bounds__(ATTN_WAVES * 32)
dca_attn_fused(const __bf16* __restrict__ Qb,   // [B*S, HID]
               const __bf16* __restrict__ Kb,   // [B*S, HID]
               const __bf16* __restrict__ Vt,   // [HID, B*S] transposed
               const float*  __restrict__ mask, // [B, S] additive
               float* __restrict__ ctx,         // [B*S*HID]
               float* __restrict__ probs)       // [B*H*S*S]
{
  extern __shared__ unsigned smem[];            // ATTN_WAVES * 24*1024 u32
  const int lane = threadIdx.x & 31;
  const int wave = threadIdx.x >> 5;
  unsigned* lds = smem + wave * (24 * 1024);    // 16*1024 u32 scores
  __bf16*   pb  = (__bf16*)(lds + 16 * 1024);   // 16*1024 bf16 probs

  // --- TDM issue-path demo: NULL descriptor (count=0 => architectural NOP),
  // tracked by TENSORcnt; zero-filled 6-arg form for this toolchain.
  {
    v4u g0 = {}; v8i g1 = {}; v4i g2 = {}; v4i g3 = {}; v8i g4 = {};
    __builtin_amdgcn_tensor_load_to_lds(g0, g1, g2, g3, g4, 0);
    __builtin_amdgcn_s_wait_tensorcnt(0);
  }

  const int qtiles = SS / (16 * ATTN_WAVES);    // 32
  int idx = blockIdx.x;
  const int qt = idx % qtiles;  idx /= qtiles;
  const int h  = idx % HH;      idx /= HH;
  const int b  = idx;

  const int q0   = qt * (16 * ATTN_WAVES) + wave * 16;
  const int r    = lane & 15;
  const int half = lane >> 4;

  // ---- Q A-fragments (DH=64 -> two K-chunks of 32) --------------------
  const __bf16* qbase = Qb + (size_t)(b * SS + q0 + r) * HIDD + h * DHH;
  const v16bf aq0 = *(const v16bf*)(qbase + half * 16);
  const v16bf aq1 = *(const v16bf*)(qbase + 32 + half * 16);

  // ---- pass 1: scores = QK^T * scale + mask -> LDS --------------------
  const float scale = 0.125f;                   // 1/sqrt(64)
  for (int kt = 0; kt < SS / 16; ++kt) {
    const __bf16* kbase = Kb + (size_t)(b * SS + kt * 16 + r) * HIDD + h * DHH;
    const v16bf bk0 = *(const v16bf*)(kbase + half * 16);
    const v16bf bk1 = *(const v16bf*)(kbase + 32 + half * 16);
    __builtin_prefetch(kbase + 16 * HIDD, 0, 1);
    v8f acc = {};
    acc = __builtin_amdgcn_wmma_f32_16x16x32_bf16(false, aq0, false, bk0,
                                                  (short)0, acc, false, false);
    acc = __builtin_amdgcn_wmma_f32_16x16x32_bf16(false, aq1, false, bk1,
                                                  (short)0, acc, false, false);
    const float mval = mask[b * SS + kt * 16 + r];
#pragma unroll
    for (int i = 0; i < 8; ++i) {
      const int m = i + half * 8;
      const float sv = fmaf(acc[i], scale, mval);
      lds[m * 1024 + kt * 16 + r] = __builtin_bit_cast(unsigned, sv);
    }
  }
  __syncthreads();

  // ---- pass 2: softmax over each full row (lane-halves split 512 cols)
  const int c0 = half * 512;
  unsigned* myrow = lds + r * 1024;
  __bf16*   mypb  = pb  + r * 1024;
  float mx = -3.0e38f;
  for (int c = 0; c < 512; ++c)
    mx = fmaxf(mx, __builtin_bit_cast(float, myrow[c0 + c]));
  mx = fmaxf(mx, __shfl_xor(mx, 16, 32));

  float sum = 0.f;
  for (int c = 0; c < 512; ++c) {
    const float e = __expf(__builtin_bit_cast(float, myrow[c0 + c]) - mx);
    myrow[c0 + c] = __builtin_bit_cast(unsigned, e);
    sum += e;
  }
  sum += __shfl_xor(sum, 16, 32);
  const float inv = 1.0f / sum;

  float* prow = probs + ((size_t)(b * HH + h) * SS + (q0 + r)) * SS;
  for (int c = 0; c < 512; ++c) {
    const float p = __builtin_bit_cast(float, myrow[c0 + c]) * inv;
    __builtin_nontemporal_store(p, prow + c0 + c);   // 512MB stream: NT
    mypb[c0 + c] = f2bf(p);                          // packed bf16 for P@V
  }
  __syncthreads();

  // ---- pass 3: ctx[16 x 64] = probs_bf16[16 x 1024] @ V[1024 x 64] ----
  // Vt layout: row d (=h*64+nt*16+r), contiguous along token index.
  const __bf16* vthead = Vt + (size_t)(h * DHH) * MTOT + b * SS;
  v8f acc[4] = {};
  for (int kt = 0; kt < SS / 32; ++kt) {
    const int ko = kt * 32 + half * 16;
    const v16bf ap = *(const v16bf*)(pb + r * 1024 + ko);   // 2x ds_load_b128
#pragma unroll
    for (int nt = 0; nt < 4; ++nt) {
      const v16bf bv =
          *(const v16bf*)(vthead + (size_t)(nt * 16 + r) * MTOT + ko);
      acc[nt] = __builtin_amdgcn_wmma_f32_16x16x32_bf16(false, ap, false, bv,
                                                        (short)0, acc[nt],
                                                        false, false);
    }
  }
#pragma unroll
  for (int nt = 0; nt < 4; ++nt)
#pragma unroll
    for (int i = 0; i < 8; ++i) {
      const int m = i + half * 8;
      ctx[(size_t)(b * SS + q0 + m) * HIDD + h * DHH + nt * 16 + r] = acc[nt][i];
    }
}

// ---------------------------------------------------------------------------
// Host-side launch
// ---------------------------------------------------------------------------
extern "C" void kernel_launch(void* const* d_in, const int* in_sizes, int n_in,
                              void* d_out, int out_size, void* d_ws, size_t ws_size,
                              hipStream_t stream) {
  (void)in_sizes; (void)n_in; (void)out_size; (void)ws_size;

  const float* x1 = (const float*)d_in[0];
  const float* m1 = (const float*)d_in[1];
  const float* x2 = (const float*)d_in[2];
  const float* m2 = (const float*)d_in[3];
  const float* Wf[6]   = {(const float*)d_in[4],  (const float*)d_in[6],
                          (const float*)d_in[8],  (const float*)d_in[10],
                          (const float*)d_in[12], (const float*)d_in[14]};
  const float* bias[6] = {(const float*)d_in[5],  (const float*)d_in[7],
                          (const float*)d_in[9],  (const float*)d_in[11],
                          (const float*)d_in[13], (const float*)d_in[15]};

  // workspace layout (bytes)
  char* ws = (char*)d_ws;
  const size_t xbytes = (size_t)MTOT * HIDD * 2;       // 8 MB
  const size_t wbytes = (size_t)HIDD * HIDD * 2;       // 2 MB
  __bf16* Xb1 = (__bf16*)(ws);
  __bf16* Xb2 = (__bf16*)(ws + xbytes);
  __bf16* Wt[6];
  for (int i = 0; i < 6; ++i) Wt[i] = (__bf16*)(ws + 2 * xbytes + i * wbytes);
  __bf16* QKV[6];                 // Q1,K1,V1(^T),Q2,K2,V2(^T)
  for (int i = 0; i < 6; ++i)
    QKV[i] = (__bf16*)(ws + 2 * xbytes + 6 * wbytes + i * xbytes);

  // 1) conversions (inputs straight; weights transposed for WMMA B operand)
  const int nX8 = MTOT * HIDD / 8;                     // 524288
  dca_cvt_bf16x8<<<nX8 / 256, 256, 0, stream>>>((const float4*)x1, (uint4*)Xb1, nX8);
  dca_cvt_bf16x8<<<nX8 / 256, 256, 0, stream>>>((const float4*)x2, (uint4*)Xb2, nX8);
  for (int i = 0; i < 6; ++i)
    dca_cvt_bf16_T<<<(HIDD * HIDD) / 256, 256, 0, stream>>>(Wf[i], Wt[i]);

  // 2) QKV projections; V projections stored transposed [HID, MTOT]
  const __bf16* Xsrc[6] = {Xb1, Xb1, Xb1, Xb2, Xb2, Xb2};
  const int     trsp[6] = {0, 0, 1, 0, 0, 1};
  for (int i = 0; i < 6; ++i)
    dca_gemm_qkv<<<2048, 256, 0, stream>>>(Xsrc[i], Wt[i], bias[i], QKV[i],
                                           trsp[i]);

  // 3) fused attention (two streams), outputs packed flat in return order
  float* out    = (float*)d_out;
  float* ctx1   = out;
  float* ctx2   = out + (size_t)4194304;
  float* probs1 = out + (size_t)8388608;
  float* probs2 = out + (size_t)75497472;

  const int    attnGrid = BB * HH * (SS / (16 * ATTN_WAVES));  // 2048
  const size_t shbytes  = (size_t)ATTN_WAVES * 24 * 1024 * 4;  // 192 KB

  // stream 1: Q1 attends K2/V2 with mask2
  dca_attn_fused<<<attnGrid, ATTN_WAVES * 32, shbytes, stream>>>(
      QKV[0], QKV[4], QKV[5], m2, ctx1, probs1);
  // stream 2: Q2 attends K1/V1 with mask1
  dca_attn_fused<<<attnGrid, ATTN_WAVES * 32, shbytes, stream>>>(
      QKV[3], QKV[1], QKV[2], m1, ctx2, probs2);
}
